// GNN_layer_88991722373412
// MI455X (gfx1250) — matrix operand order repaired and checked by reading, not compiled, not run on hardware
//
#include <hip/hip_runtime.h>
#include <hip/hip_bf16.h>

typedef __attribute__((ext_vector_type(16))) __bf16 v16bf;
typedef __attribute__((ext_vector_type(8)))  __bf16 v8bf;
typedef __attribute__((ext_vector_type(8)))  float  v8f;
typedef __attribute__((ext_vector_type(4)))  float  v4f;

namespace {
constexpr int NUE   = 16;    // nodes (leave-one-out axis)
constexpr int BATCH = 4096;
constexpr int DIM   = 512;
constexpr int KC    = 256;          // K-chunk staged in LDS
constexpr int DSTR  = KC + 8;       // padded bf16 row stride (16B shift/row -> no bank conflicts)
constexpr int KSTEPS = KC / 32;     // WMMA K-steps per chunk
constexpr int SMEM_BYTES = 4 * 16 * DSTR * 2;  // 33792 B (aliases 16KB prod buffer)
}

// One block = 16 waves = one wave per node n, all on the same (b-tile, e-tile).
// Each wave: 4 accumulators (m1..m4), 64 bf16 WMMAs, software-pipelined A/B frags.
__global__ __launch_bounds__(512) void gnn_fused_wmma(
    const float* __restrict__ x,
    const float* __restrict__ W1, const float* __restrict__ b1,
    const float* __restrict__ W2, const float* __restrict__ b2,
    const float* __restrict__ W3, const float* __restrict__ b3,
    const float* __restrict__ W4, const float* __restrict__ b4,
    float* __restrict__ out)
{
  __shared__ __align__(32) unsigned char smem[SMEM_BYTES];
  __bf16 (*wlds)[16][DSTR] = reinterpret_cast<__bf16 (*)[16][DSTR]>(smem); // [mat][e][k]
  float  (*plds)[32][8]    = reinterpret_cast<float  (*)[32][8]>(smem);    // [n][lane][r]

  const int tid  = threadIdx.x;
  const int lane = tid & 31;
  const int wv   = tid >> 5;          // node index n = 0..15
  const int half = lane >> 4;         // lane half (ISA 16-bit A/B layouts)
  const int mrow = lane & 15;         // A row (b) / B column (e) within tile

  const int e0 = blockIdx.x * 16;     // e-tile fast-varying -> x slice reused via L2
  const int b0 = blockIdx.y * 16;

  const float* const Wp[4] = {W1, W2, W3, W4};

  const float bias0 = b1[e0 + mrow];
  const float bias1 = b2[e0 + mrow];
  const float bias2 = b3[e0 + mrow];
  const float bias3 = b4[e0 + mrow];

  const float* xrow = x + ((size_t)wv * BATCH + (size_t)(b0 + mrow)) * DIM;

  // A fragment loader: 16-bit A 16x32 layout -> lane holds K runs
  // [half*8 .. +7] and [16+half*8 .. +7] of row mrow. kg = absolute K base.
  auto loadA = [&](int kg) -> v16bf {
    const int klo = kg + half * 8;
    const int khi = klo + 16;
    v4f f0 = *(const v4f*)(xrow + klo);
    v4f f1 = *(const v4f*)(xrow + klo + 4);
    v4f f2 = *(const v4f*)(xrow + khi);
    v4f f3 = *(const v4f*)(xrow + khi + 4);
    v16bf a;
#pragma unroll
    for (int j = 0; j < 4; ++j) {
      a[j]      = (__bf16)f0[j];
      a[4 + j]  = (__bf16)f1[j];
      a[8 + j]  = (__bf16)f2[j];
      a[12 + j] = (__bf16)f3[j];
    }
    return a;
  };

  // B fragment loader: 16-bit B 32x16 layout -> lane holds K runs
  // [half*8 .. +7] and [16+half*8 .. +7] of column mrow. ksl = K-step index within chunk.
  auto loadB = [&](int i, int ksl) -> v16bf {
    const __bf16* wp = &wlds[i][mrow][0] + ksl * 32 + half * 8;
    v8bf lo = *(const v8bf*)(wp);
    v8bf hi = *(const v8bf*)(wp + 16);
    v16bf bm;
#pragma unroll
    for (int j = 0; j < 8; ++j) { bm[j] = lo[j]; bm[8 + j] = hi[j]; }
    return bm;
  };

  v8f acc[4];
#pragma unroll
  for (int i = 0; i < 4; ++i)
    acc[i] = (v8f){0.f, 0.f, 0.f, 0.f, 0.f, 0.f, 0.f, 0.f};

  for (int c = 0; c < DIM / KC; ++c) {
    const int kbase = c * KC;

    // ---- stage W1..W4 K-chunk into LDS as bf16: 8 elems/thread/iter ----
    // (2x global_load_b128 + 4x v_cvt_pk_bf16_f32 + 1x ds_store_b128)
#pragma unroll
    for (int t8 = tid; t8 < 4 * 16 * KC / 8; t8 += 512) {
      const int i = t8 >> 9;            // / (16*KC/8) == 512
      const int r = t8 & 511;
      const int e = r >> 5;             // KC/8 == 32 octets per row
      const int d = (r & 31) * 8;
      const float* src = Wp[i] + (size_t)(e0 + e) * DIM + (kbase + d);
      v4f f0 = *(const v4f*)(src);
      v4f f1 = *(const v4f*)(src + 4);
      v8bf pk;
#pragma unroll
      for (int j = 0; j < 4; ++j) {
        pk[j]     = (__bf16)f0[j];
        pk[4 + j] = (__bf16)f1[j];
      }
      *(v8bf*)(&wlds[i][e][d]) = pk;
    }
    __syncthreads();

    // ---- software-pipelined K loop ----
    v16bf a = loadA(kbase);
    v16bf b[4];
#pragma unroll
    for (int i = 0; i < 4; ++i) b[i] = loadB(i, 0);

#pragma unroll
    for (int ks = 0; ks < KSTEPS; ++ks) {
      v16bf an;
      v16bf bn[4];
      if (ks + 1 < KSTEPS) {            // prefetch next step into fresh regs
        an = loadA(kbase + (ks + 1) * 32);
#pragma unroll
        for (int i = 0; i < 4; ++i) bn[i] = loadB(i, ks + 1);
      }
#pragma unroll
      for (int i = 0; i < 4; ++i)
        acc[i] = __builtin_amdgcn_wmma_f32_16x16x32_bf16(
            false, a, false, b[i], (short)0, acc[i], false, false);
      if (ks + 1 < KSTEPS) {
        a = an;
#pragma unroll
        for (int i = 0; i < 4; ++i) b[i] = bn[i];
      }
    }

    if (c + 1 < DIM / KC) __syncthreads();   // chunk consumed before restage
  }

  // ---- epilogue: bias, prod = m2*m3, cross-n LDS reduction, mish ----
  v8f prod;
#pragma unroll
  for (int r = 0; r < 8; ++r)
    prod[r] = (acc[1][r] + bias1) * (acc[2][r] + bias2);

  __syncthreads();                    // W LDS dead -> reuse for prod
  {
    float* pp = &plds[wv][lane][0];
#pragma unroll
    for (int r = 0; r < 8; ++r) pp[r] = prod[r];
  }
  __syncthreads();

  v8f tot = (v8f){0.f, 0.f, 0.f, 0.f, 0.f, 0.f, 0.f, 0.f};
#pragma unroll
  for (int nn = 0; nn < NUE; ++nn) {
    const float* q = &plds[nn][lane][0];
    v4f t0 = *(const v4f*)q;
    v4f t1 = *(const v4f*)(q + 4);
#pragma unroll
    for (int j = 0; j < 4; ++j) { tot[j] += t0[j]; tot[4 + j] += t1[j]; }
  }

  // C/D layout: N = mrow, M = r + 8*half
  const float invn = 1.0f / (float)(NUE - 1);
  float* obase = out + ((size_t)wv * BATCH + (size_t)(b0 + half * 8)) * DIM
                     + (e0 + mrow);
#pragma unroll
  for (int r = 0; r < 8; ++r) {
    const float m1v = acc[0][r] + bias0;
    const float m4v = acc[3][r] + bias3;
    const float me  = (tot[r] - prod[r]) * invn;
    const float v   = m1v + m4v * me;
    const float sp  = __logf(1.0f + __expf(v));   // softplus
    obase[(size_t)r * DIM] = v * tanhf(sp);       // mish
  }
}

extern "C" void kernel_launch(void* const* d_in, const int* in_sizes, int n_in,
                              void* d_out, int out_size, void* d_ws, size_t ws_size,
                              hipStream_t stream) {
  (void)in_sizes; (void)n_in; (void)out_size; (void)d_ws; (void)ws_size;
  const float* x  = (const float*)d_in[0];
  const float* W1 = (const float*)d_in[1];
  const float* b1 = (const float*)d_in[2];
  const float* W2 = (const float*)d_in[3];
  const float* b2 = (const float*)d_in[4];
  const float* W3 = (const float*)d_in[5];
  const float* b3 = (const float*)d_in[6];
  const float* W4 = (const float*)d_in[7];
  const float* b4 = (const float*)d_in[8];
  float* out = (float*)d_out;

  dim3 grid(DIM / 16, BATCH / 16);    // (32 e-tiles, 256 b-tiles)
  dim3 block(512);                    // 16 waves: one per node n
  gnn_fused_wmma<<<grid, block, 0, stream>>>(x, W1, b1, W2, b2, W3, b3, W4, b4, out);
}
